// RGCN_50044958933136
// MI455X (gfx1250) — matrix operand order repaired and checked by reading, not compiled, not run on hardware
//
#include <hip/hip_runtime.h>

#define P_N   100000
#define A_N   50000
#define IN_C  128
#define H_C   256
#define OUT_C 349
#define OUT_P 384
#define E_WR  600000
#define E_CI  1000000

typedef __attribute__((ext_vector_type(8)))  float  v8f;
typedef __attribute__((ext_vector_type(16))) __bf16 v16bf;
typedef __attribute__((ext_vector_type(8)))  __bf16 v8bf;

// ---------------------------------------------------------------- CDNA5 async copy helpers
// GLOBAL_LOAD_ASYNC_TO_LDS_B128 (GVS mode): saddr = 64-bit base, vaddr = 32-bit offset,
// vdst = LDS byte address.  Tracked by ASYNCcnt.
__device__ __forceinline__ void async_ld_b128(unsigned lds_addr, unsigned goff,
                                              unsigned long long base) {
  asm volatile("global_load_async_to_lds_b128 %0, %1, %2"
               :: "v"(lds_addr), "v"(goff), "s"(base)
               : "memory");
}
__device__ __forceinline__ void wait_async0() {
  asm volatile("s_wait_asynccnt 0" ::: "memory");
}

// ---------------------------------------------------------------- utility
__global__ __launch_bounds__(256) void k_convert_f32_bf16(const float* __restrict__ in,
                                                          __bf16* __restrict__ out, long n) {
  long i = (long)blockIdx.x * blockDim.x + threadIdx.x;
  if (i < n) out[i] = (__bf16)in[i];
}

__global__ __launch_bounds__(256) void k_zero_f32(float* __restrict__ p, long n) {
  long i = (long)blockIdx.x * blockDim.x + threadIdx.x;
  if (i < n) p[i] = 0.0f;
}

// W [K][N] fp32 row-major  ->  WT [Npad][K] bf16 row-major (zero pad n>=N)
__global__ __launch_bounds__(256) void k_transpose_w(const float* __restrict__ W,
                                                     __bf16* __restrict__ WT,
                                                     int K, int N, int Npad) {
  long i = (long)blockIdx.x * blockDim.x + threadIdx.x;
  if (i >= (long)Npad * K) return;
  int n = (int)(i / K), k = (int)(i % K);
  WT[i] = (n < N) ? (__bf16)W[(size_t)k * N + n] : (__bf16)0.0f;
}

__global__ __launch_bounds__(256) void k_pad_bias(const float* __restrict__ b,
                                                  float* __restrict__ out, int N, int Npad) {
  int i = blockIdx.x * blockDim.x + threadIdx.x;
  if (i < Npad) out[i] = (i < N) ? b[i] : 0.0f;
}

// ---------------------------------------------------------------- aggregation
__global__ __launch_bounds__(256) void k_count_deg(const int* __restrict__ dst, int E,
                                                   float* __restrict__ deg) {
  int e = blockIdx.x * blockDim.x + threadIdx.x;
  if (e < E) atomicAdd(&deg[dst[e]], 1.0f);
}

// feat rows are H_C=256 bf16; 32 lanes * 8ch per edge
__global__ __launch_bounds__(256) void k_scatter_add(const __bf16* __restrict__ feat,
                                                     const int* __restrict__ src,
                                                     const int* __restrict__ dst,
                                                     int E, float* __restrict__ accum) {
  long idx = (long)blockIdx.x * blockDim.x + threadIdx.x;
  long e = idx >> 5;
  if (e >= E) return;
  int ch = (int)(idx & 31) * 8;
  int s = src[e], d = dst[e];
  v8bf v = *reinterpret_cast<const v8bf*>(feat + (size_t)s * H_C + ch);
  float* ap = accum + (size_t)d * H_C + ch;
#pragma unroll
  for (int i = 0; i < 8; ++i) atomicAdd(ap + i, (float)v[i]);
}

__global__ __launch_bounds__(256) void k_finalize_mean(const float* __restrict__ accum,
                                                       const float* __restrict__ deg,
                                                       __bf16* __restrict__ out, long n) {
  long i = (long)blockIdx.x * blockDim.x + threadIdx.x;
  if (i >= n) return;
  int node = (int)(i >> 8);  // 256 channels per node
  float inv = 1.0f / fmaxf(deg[node], 1.0f);
  out[i] = (__bf16)(accum[i] * inv);
}

// ---------------------------------------------------------------- fused WMMA GEMM
// out[M,N] = relu?( sum_p A_p[M,K] @ W_p^T + bias )  (bf16 or fp32 store)
// block: 256 threads = 8 waves; tile 128(M) x 64(N); K step 64; async global->LDS staging.
template <int NPAIRS, bool OUTF32, bool RELU>
__global__ __launch_bounds__(256) void k_gemm_bf16(
    const __bf16* __restrict__ A0, const __bf16* __restrict__ A1, const __bf16* __restrict__ A2,
    const __bf16* __restrict__ W0, const __bf16* __restrict__ W1, const __bf16* __restrict__ W2,
    const float* __restrict__ bias, int M, int K,
    __bf16* __restrict__ outBf, float* __restrict__ outF, int Nstore) {
  __shared__ uint4 ldsA_raw[1024];  // 128 x 64 bf16 = 16 KB
  __shared__ uint4 ldsB_raw[512];   //  64 x 64 bf16 =  8 KB
  __bf16* ldsA = reinterpret_cast<__bf16*>(ldsA_raw);
  __bf16* ldsB = reinterpret_cast<__bf16*>(ldsB_raw);

  const int tid  = threadIdx.x;
  const int wave = tid >> 5;
  const int lane = tid & 31;
  const int half = lane >> 4;
  const int l16  = lane & 15;
  const int mBase   = blockIdx.y * 128;
  const int rowBase = mBase + wave * 16;
  const int nBase   = blockIdx.x * 64;

  v8f acc[4] = {};

  const __bf16* Aps[3] = {A0, A1, A2};
  const __bf16* Wps[3] = {W0, W1, W2};

#pragma unroll
  for (int p = 0; p < NPAIRS; ++p) {
    const unsigned long long Ap = (unsigned long long)(uintptr_t)Aps[p];
    const unsigned long long Wp = (unsigned long long)(uintptr_t)Wps[p];
    for (int k0 = 0; k0 < K; k0 += 64) {
      __syncthreads();  // previous stage's LDS reads complete
      // ---- A tile: 128 rows x 64 bf16 (128B/row); each thread 4 async b128
      {
        int r = tid >> 1, part = tid & 1;
        int grow = mBase + r;
        if (grow < M) {  // EXEC-masked async issue; stale LDS rows masked at store
          unsigned gbase = (unsigned)(((size_t)grow * K + k0 + part * 32) * 2);
          unsigned lbase = (unsigned)(uintptr_t)(ldsA + r * 64 + part * 32);
#pragma unroll
          for (int c = 0; c < 4; ++c)
            async_ld_b128(lbase + c * 16, gbase + c * 16, Ap);
        }
      }
      // ---- B tile: 64 rows x 64 bf16; each thread 2 async b128 (no guard: Npad mult of 64)
      {
        int nl = tid >> 2, q = tid & 3;
        unsigned gbase = (unsigned)(((size_t)(nBase + nl) * K + k0 + q * 16) * 2);
        unsigned lbase = (unsigned)(uintptr_t)(ldsB + nl * 64 + q * 16);
        async_ld_b128(lbase, gbase, Wp);
        async_ld_b128(lbase + 16, gbase + 16, Wp);
      }
      wait_async0();
      __syncthreads();

      // ---- compute: 2 K-halves x 4 N-tiles = 8 WMMA / wave / stage
#pragma unroll
      for (int h2 = 0; h2 < 2; ++h2) {
        const __bf16* arow = ldsA + (wave * 16 + l16) * 64 + h2 * 32;
        v8bf aLo = *reinterpret_cast<const v8bf*>(arow + half * 16);
        v8bf aHi = *reinterpret_cast<const v8bf*>(arow + half * 16 + 8);
        v16bf afrag = __builtin_shufflevector(aLo, aHi, 0, 1, 2, 3, 4, 5, 6, 7,
                                              8, 9, 10, 11, 12, 13, 14, 15);
#pragma unroll
        for (int j = 0; j < 4; ++j) {
          const __bf16* brow = ldsB + (j * 16 + l16) * 64 + h2 * 32 + half * 16;
          v8bf bLo = *reinterpret_cast<const v8bf*>(brow);
          v8bf bHi = *reinterpret_cast<const v8bf*>(brow + 8);
          v16bf bfrag = __builtin_shufflevector(bLo, bHi, 0, 1, 2, 3, 4, 5, 6, 7,
                                                8, 9, 10, 11, 12, 13, 14, 15);
          acc[j] = __builtin_amdgcn_wmma_f32_16x16x32_bf16(
              false, afrag, false, bfrag, (short)0, acc[j], false, false);
        }
      }
    }
  }

  // store: VGPR r -> M = half*8 + r ; N = lane%16
#pragma unroll
  for (int j = 0; j < 4; ++j) {
    int col = nBase + j * 16 + l16;
    float bv = bias[col];
#pragma unroll
    for (int r = 0; r < 8; ++r) {
      int row = rowBase + half * 8 + r;
      if (row >= M) continue;
      float v = acc[j][r] + bv;
      if (RELU) v = fmaxf(v, 0.0f);
      if (OUTF32) {
        if (col < Nstore) outF[(size_t)row * Nstore + col] = v;
      } else {
        if (col < Nstore) outBf[(size_t)row * Nstore + col] = (__bf16)v;
      }
    }
  }
}

// ---------------------------------------------------------------- log_softmax (wave per row)
__global__ __launch_bounds__(256) void k_log_softmax(float* __restrict__ out, int M, int N) {
  int wave = threadIdx.x >> 5;
  int lane = threadIdx.x & 31;
  int row = blockIdx.x * 8 + wave;
  if (row >= M) return;
  float vals[11];
  float mx = -3.4e38f;
#pragma unroll
  for (int i = 0; i < 11; ++i) {
    int c = lane + i * 32;
    vals[i] = (c < N) ? out[(size_t)row * N + c] : -3.4e38f;
    mx = fmaxf(mx, vals[i]);
  }
#pragma unroll
  for (int off = 16; off > 0; off >>= 1) mx = fmaxf(mx, __shfl_xor(mx, off, 32));
  float s = 0.0f;
#pragma unroll
  for (int i = 0; i < 11; ++i) {
    int c = lane + i * 32;
    if (c < N) s += __expf(vals[i] - mx);
  }
#pragma unroll
  for (int off = 16; off > 0; off >>= 1) s += __shfl_xor(s, off, 32);
  float lg = __logf(s);
#pragma unroll
  for (int i = 0; i < 11; ++i) {
    int c = lane + i * 32;
    if (c < N) out[(size_t)row * N + c] = vals[i] - mx - lg;
  }
}

// ---------------------------------------------------------------- host
extern "C" void kernel_launch(void* const* d_in, const int* in_sizes, int n_in,
                              void* d_out, int out_size, void* d_ws, size_t ws_size,
                              hipStream_t stream) {
  (void)in_sizes; (void)n_in; (void)out_size; (void)ws_size;
  const float* x_paper    = (const float*)d_in[0];
  const float* emb_author = (const float*)d_in[1];
  const float* lin_w      = (const float*)d_in[2];
  const float* lin_b      = (const float*)d_in[3];
  const float* c0rp_w     = (const float*)d_in[4];
  const float* c0rp_b     = (const float*)d_in[5];
  const float* c0ra_w     = (const float*)d_in[6];
  const float* c0ra_b     = (const float*)d_in[7];
  const float* c0w_w      = (const float*)d_in[8];
  const float* c0c_w      = (const float*)d_in[9];
  const float* c0r_w      = (const float*)d_in[10];
  const float* c1r_w      = (const float*)d_in[11];
  const float* c1r_b      = (const float*)d_in[12];
  const float* c1w_w      = (const float*)d_in[13];
  const float* c1c_w      = (const float*)d_in[14];
  const int* wr_src = (const int*)d_in[15];
  const int* wr_dst = (const int*)d_in[16];
  const int* ci_src = (const int*)d_in[17];
  const int* ci_dst = (const int*)d_in[18];
  const int* rv_src = (const int*)d_in[19];
  const int* rv_dst = (const int*)d_in[20];
  float* logits = (float*)d_out;

  char* ws = (char*)d_ws;
  size_t off = 0;
  auto alloc = [&](size_t bytes) -> char* {
    char* p = ws + off;
    off = (off + bytes + 255) & ~(size_t)255;
    return p;
  };
  __bf16* hp_bf   = (__bf16*)alloc((size_t)P_N * H_C * 2);
  __bf16* ha_bf   = (__bf16*)alloc((size_t)A_N * H_C * 2);
  __bf16* xp_bf   = (__bf16*)alloc((size_t)P_N * IN_C * 2);
  __bf16* aggw_bf = (__bf16*)alloc((size_t)P_N * H_C * 2);
  __bf16* aggc_bf = (__bf16*)alloc((size_t)P_N * H_C * 2);
  __bf16* aggr_bf = (__bf16*)alloc((size_t)A_N * H_C * 2);
  __bf16* hp1_bf  = (__bf16*)alloc((size_t)P_N * H_C * 2);
  __bf16* ha1_bf  = (__bf16*)alloc((size_t)A_N * H_C * 2);
  float*  accum   = (float*) alloc((size_t)P_N * H_C * 4);
  float*  deg_w   = (float*) alloc((size_t)P_N * 4);
  float*  deg_c   = (float*) alloc((size_t)P_N * 4);
  float*  deg_r   = (float*) alloc((size_t)A_N * 4);
  __bf16* WT_lin  = (__bf16*)alloc((size_t)H_C * IN_C * 2);
  __bf16* WT_c0rp = (__bf16*)alloc((size_t)H_C * H_C * 2);
  __bf16* WT_c0ra = (__bf16*)alloc((size_t)H_C * H_C * 2);
  __bf16* WT_c0w  = (__bf16*)alloc((size_t)H_C * H_C * 2);
  __bf16* WT_c0c  = (__bf16*)alloc((size_t)H_C * H_C * 2);
  __bf16* WT_c0r  = (__bf16*)alloc((size_t)H_C * H_C * 2);
  __bf16* WT_c1r  = (__bf16*)alloc((size_t)OUT_P * H_C * 2);
  __bf16* WT_c1w  = (__bf16*)alloc((size_t)OUT_P * H_C * 2);
  __bf16* WT_c1c  = (__bf16*)alloc((size_t)OUT_P * H_C * 2);
  float*  b_c1    = (float*) alloc((size_t)OUT_P * 4);

  auto blocks = [](long n) { return (int)((n + 255) / 256); };

  // ---- prep: conversions, weight transposes, bias pad
  k_convert_f32_bf16<<<blocks((long)P_N * IN_C), 256, 0, stream>>>(x_paper, xp_bf, (long)P_N * IN_C);
  k_convert_f32_bf16<<<blocks((long)A_N * H_C), 256, 0, stream>>>(emb_author, ha_bf, (long)A_N * H_C);
  k_transpose_w<<<blocks((long)H_C * IN_C), 256, 0, stream>>>(lin_w, WT_lin, IN_C, H_C, H_C);
  k_transpose_w<<<blocks((long)H_C * H_C), 256, 0, stream>>>(c0rp_w, WT_c0rp, H_C, H_C, H_C);
  k_transpose_w<<<blocks((long)H_C * H_C), 256, 0, stream>>>(c0ra_w, WT_c0ra, H_C, H_C, H_C);
  k_transpose_w<<<blocks((long)H_C * H_C), 256, 0, stream>>>(c0w_w, WT_c0w, H_C, H_C, H_C);
  k_transpose_w<<<blocks((long)H_C * H_C), 256, 0, stream>>>(c0c_w, WT_c0c, H_C, H_C, H_C);
  k_transpose_w<<<blocks((long)H_C * H_C), 256, 0, stream>>>(c0r_w, WT_c0r, H_C, H_C, H_C);
  k_transpose_w<<<blocks((long)OUT_P * H_C), 256, 0, stream>>>(c1r_w, WT_c1r, H_C, OUT_C, OUT_P);
  k_transpose_w<<<blocks((long)OUT_P * H_C), 256, 0, stream>>>(c1w_w, WT_c1w, H_C, OUT_C, OUT_P);
  k_transpose_w<<<blocks((long)OUT_P * H_C), 256, 0, stream>>>(c1c_w, WT_c1c, H_C, OUT_C, OUT_P);
  k_pad_bias<<<blocks(OUT_P), 256, 0, stream>>>(c1r_b, b_c1, OUT_C, OUT_P);

  // ---- degrees (shared across layers)
  k_zero_f32<<<blocks(P_N), 256, 0, stream>>>(deg_w, P_N);
  k_zero_f32<<<blocks(P_N), 256, 0, stream>>>(deg_c, P_N);
  k_zero_f32<<<blocks(A_N), 256, 0, stream>>>(deg_r, A_N);
  k_count_deg<<<blocks(E_WR), 256, 0, stream>>>(wr_dst, E_WR, deg_w);
  k_count_deg<<<blocks(E_CI), 256, 0, stream>>>(ci_dst, E_CI, deg_c);
  k_count_deg<<<blocks(E_WR), 256, 0, stream>>>(rv_dst, E_WR, deg_r);

  auto aggregate = [&](const __bf16* feat, const int* src, const int* dst, int E,
                       const float* deg, int nodes, __bf16* outbf) {
    long n = (long)nodes * H_C;
    k_zero_f32<<<blocks(n), 256, 0, stream>>>(accum, n);
    k_scatter_add<<<blocks((long)E * 32), 256, 0, stream>>>(feat, src, dst, E, accum);
    k_finalize_mean<<<blocks(n), 256, 0, stream>>>(accum, deg, outbf, n);
  };

  // ---- input projection: h_p = x_paper @ lin_w + lin_b
  {
    dim3 grid(H_C / 64, (P_N + 127) / 128);
    k_gemm_bf16<1, false, false><<<grid, 256, 0, stream>>>(
        xp_bf, nullptr, nullptr, WT_lin, nullptr, nullptr, lin_b, P_N, IN_C,
        hp_bf, nullptr, H_C);
  }

  // ---- layer-0 aggregations
  aggregate(ha_bf, wr_src, wr_dst, E_WR, deg_w, P_N, aggw_bf);
  aggregate(hp_bf, ci_src, ci_dst, E_CI, deg_c, P_N, aggc_bf);
  aggregate(hp_bf, rv_src, rv_dst, E_WR, deg_r, A_N, aggr_bf);

  // ---- layer-0 fused GEMMs (+bias, relu)
  {
    dim3 grid(H_C / 64, (P_N + 127) / 128);
    k_gemm_bf16<3, false, true><<<grid, 256, 0, stream>>>(
        hp_bf, aggw_bf, aggc_bf, WT_c0rp, WT_c0w, WT_c0c, c0rp_b, P_N, H_C,
        hp1_bf, nullptr, H_C);
  }
  {
    dim3 grid(H_C / 64, (A_N + 127) / 128);
    k_gemm_bf16<2, false, true><<<grid, 256, 0, stream>>>(
        ha_bf, aggr_bf, nullptr, WT_c0ra, WT_c0r, nullptr, c0ra_b, A_N, H_C,
        ha1_bf, nullptr, H_C);
  }

  // ---- layer-1 aggregations (same edges, new features)
  aggregate(ha1_bf, wr_src, wr_dst, E_WR, deg_w, P_N, aggw_bf);
  aggregate(hp1_bf, ci_src, ci_dst, E_CI, deg_c, P_N, aggc_bf);

  // ---- final fused GEMM -> fp32 logits [P, 349]
  {
    dim3 grid(OUT_P / 64, (P_N + 127) / 128);
    k_gemm_bf16<3, true, false><<<grid, 256, 0, stream>>>(
        hp1_bf, aggw_bf, aggc_bf, WT_c1r, WT_c1w, WT_c1c, b_c1, P_N, H_C,
        nullptr, logits, OUT_C);
  }

  // ---- log_softmax in place
  k_log_softmax<<<(P_N + 7) / 8, 256, 0, stream>>>(logits, P_N, OUT_C);
}
